// RNN_7249904796236
// MI455X (gfx1250) — matrix operand order repaired and checked by reading, not compiled, not run on hardware
//
#include <hip/hip_runtime.h>
#include <hip/hip_bf16.h>

// ---------------------------------------------------------------------------
// tanh-RNN (512 sequential steps) + classifier on gfx1250.
// Strategy: one kernel launch per scan step; each step is a bf16-WMMA GEMM
//   h_new = tanh( concat(x_t, h) @ W^T + b )        [256 x 1024], K = 1536
// followed by a final bf16-WMMA classifier GEMM [256x1024]x[1024x1000].
// W / Wc are converted fp32 -> bf16 once per launch (W stays L2-resident,
// 3 MB << 192 MB L2). The concat A-tile (16 rows x 1536) is staged into LDS
// as bf16 and shared by all 8 waves of a block.
// ---------------------------------------------------------------------------

typedef __attribute__((ext_vector_type(16))) __bf16 v16bf;
typedef __attribute__((ext_vector_type(8)))  __bf16 v8bf;
typedef __attribute__((ext_vector_type(8)))  float  v8f;
typedef __attribute__((ext_vector_type(4)))  float  v4f;

#define SEQ   512
#define BATCH 256
#define DIN   512
#define HID   1024
#define KCAT  (DIN + HID)   // 1536
#define NCLS  1000
#define NCLSP 1008          // padded to multiple of 16
#define ROWP  1544          // LDS row pitch (bf16): 1536 + 8 -> bank-conflict-free

// ---- fp32 -> bf16 convert -------------------------------------------------
__global__ __launch_bounds__(256) void cvt_f32_bf16(const float* __restrict__ src,
                                                    __bf16* __restrict__ dst, int n) {
  int i = blockIdx.x * 256 + threadIdx.x;
  if (i < n) dst[i] = (__bf16)src[i];
}

// Wc [1000][1024] -> bf16 padded [1008][1024] (pad rows are zero)
__global__ __launch_bounds__(256) void cvt_wc(const float* __restrict__ Wc,
                                              __bf16* __restrict__ dst) {
  int i = blockIdx.x * 256 + threadIdx.x;   // over 1008*1024
  int j = i >> 10;
  int k = i & 1023;
  dst[i] = (j < NCLS) ? (__bf16)Wc[j * HID + k] : (__bf16)0.0f;
}

__global__ __launch_bounds__(256) void zero_f32(float* __restrict__ p, int n) {
  int i = blockIdx.x * 256 + threadIdx.x;
  if (i < n) p[i] = 0.0f;
}

// ---- one RNN step: h_new = tanh([x_t, h] @ W^T + b) -----------------------
// Grid: 64 blocks (16 M-tiles x 4 N-groups), 256 threads = 8 waves.
// Each wave computes 2 adjacent 16x16 output tiles (independent WMMA chains).
__global__ __launch_bounds__(256) void rnn_step(
    const float*  __restrict__ xt,     // [256][512]
    const float*  __restrict__ hprev,  // [256][1024]
    const __bf16* __restrict__ Wb,     // [1024][1536] bf16, row-major (j,k)
    const float*  __restrict__ bias,   // [1024]
    float*        __restrict__ hnew)   // [256][1024]
{
  __shared__ __bf16 Atile[16 * ROWP];

  const int tid    = threadIdx.x;
  const int mtile  = blockIdx.x >> 2;   // 0..15  (batch rows /16)
  const int ngroup = blockIdx.x & 3;    // 0..3   (256 output cols each)
  const int wave   = tid >> 5;
  const int lane   = tid & 31;

  // Stage A = concat(x_t_rows, h_rows) for this M-tile into LDS as bf16.
  for (int m = 0; m < 16; ++m) {
    const int    row = mtile * 16 + m;
    const float* xr  = xt    + (size_t)row * DIN;
    const float* hr  = hprev + (size_t)row * HID;
    for (int k = tid; k < KCAT; k += 256) {
      float v = (k < DIN) ? xr[k] : hr[k - DIN];
      Atile[m * ROWP + k] = (__bf16)v;
    }
  }
  __syncthreads();

  // Per-lane WMMA layout decode (16x16x32 bf16, wave32):
  const int m  = lane & 15;              // A: M = lane%16
  const int kA = (lane >> 4) * 16;       // A: K-base 0 / 16
  const int n  = lane & 15;              // B: N = lane%16
  const int kB = (lane >> 4) * 16;       // B: K-base 0 / 16
  const int nt0 = ngroup * 16 + wave * 2;   // first of 2 N-tiles for this wave

  const __bf16* arow  = Atile + m * ROWP;
  const __bf16* brow0 = Wb + (size_t)(nt0 * 16 + n) * KCAT;
  const __bf16* brow1 = Wb + (size_t)((nt0 + 1) * 16 + n) * KCAT;

  v8f acc0 = {};
  v8f acc1 = {};

  for (int k0 = 0; k0 < KCAT; k0 += 32) {
    // A: elements 0..15 = A[m][k0+kA .. +15]
    v8bf alo = *(const v8bf*)(arow + k0 + kA);
    v8bf ahi = *(const v8bf*)(arow + k0 + kA + 8);
    v16bf a = __builtin_shufflevector(alo, ahi,
                                      0, 1, 2, 3, 4, 5, 6, 7,
                                      8, 9, 10, 11, 12, 13, 14, 15);
    // B: 16 contiguous bf16 of W row (= column of B) starting at k0+kB
    v16bf b0 = *(const v16bf*)(brow0 + k0 + kB);
    v16bf b1 = *(const v16bf*)(brow1 + k0 + kB);

    acc0 = __builtin_amdgcn_wmma_f32_16x16x32_bf16(false, a, false, b0,
                                                   (short)0, acc0, false, false);
    acc1 = __builtin_amdgcn_wmma_f32_16x16x32_bf16(false, a, false, b1,
                                                   (short)0, acc1, false, false);
  }

  // Epilogue: bias + tanh, write f32 hidden state.
  // C/D layout: lane holds N=lane%16, rows M = v + 8*(lane>=16), v=0..7.
  const int mrow = (lane >> 4) * 8;
  {
    int col = nt0 * 16 + n;
    float bb = bias[col];
    #pragma unroll
    for (int v = 0; v < 8; ++v) {
      int row = mtile * 16 + mrow + v;
      hnew[(size_t)row * HID + col] = tanhf(acc0[v] + bb);
    }
  }
  {
    int col = (nt0 + 1) * 16 + n;
    float bb = bias[col];
    #pragma unroll
    for (int v = 0; v < 8; ++v) {
      int row = mtile * 16 + mrow + v;
      hnew[(size_t)row * HID + col] = tanhf(acc1[v] + bb);
    }
  }
}

// ---- classifier: out = h_final @ Wc^T + bc  (N padded 1000 -> 1008) -------
// 1008 wave-tiles = 16 M-tiles x 63 N-tiles; 126 blocks x 8 waves, 1 tile/wave.
__global__ __launch_bounds__(256) void classifier(
    const float*  __restrict__ h,     // [256][1024]
    const __bf16* __restrict__ Wcb,   // [1008][1024] bf16 (padded)
    const float*  __restrict__ bc,    // [1000]
    float*        __restrict__ out)   // [256][1000]
{
  const int tid  = threadIdx.x;
  const int lane = tid & 31;
  const int gw   = blockIdx.x * 8 + (tid >> 5);  // 0..1007
  const int mtile = gw / 63;
  const int ntile = gw % 63;

  const int m  = lane & 15;
  const int kA = (lane >> 4) * 16;
  const int n  = lane & 15;
  const int kB = (lane >> 4) * 16;

  const float*  arow = h + (size_t)(mtile * 16 + m) * HID;
  const __bf16* brow = Wcb + (size_t)(ntile * 16 + n) * HID;

  v8f acc = {};
  for (int k0 = 0; k0 < HID; k0 += 32) {
    // Convert A rows fp32 -> bf16 on the fly (one 16-wide contiguous run).
    v4f f0 = *(const v4f*)(arow + k0 + kA);
    v4f f1 = *(const v4f*)(arow + k0 + kA + 4);
    v4f f2 = *(const v4f*)(arow + k0 + kA + 8);
    v4f f3 = *(const v4f*)(arow + k0 + kA + 12);
    v16bf a{};
    #pragma unroll
    for (int i = 0; i < 4; ++i) {
      a[i]      = (__bf16)f0[i];
      a[4 + i]  = (__bf16)f1[i];
      a[8 + i]  = (__bf16)f2[i];
      a[12 + i] = (__bf16)f3[i];
    }
    v16bf b = *(const v16bf*)(brow + k0 + kB);
    acc = __builtin_amdgcn_wmma_f32_16x16x32_bf16(false, a, false, b,
                                                  (short)0, acc, false, false);
  }

  const int col = ntile * 16 + n;
  if (col < NCLS) {
    const float bb   = bc[col];
    const int   mrow = (lane >> 4) * 8;
    #pragma unroll
    for (int v = 0; v < 8; ++v) {
      int row = mtile * 16 + mrow + v;
      out[(size_t)row * NCLS + col] = acc[v] + bb;
    }
  }
}

// ---------------------------------------------------------------------------
extern "C" void kernel_launch(void* const* d_in, const int* in_sizes, int n_in,
                              void* d_out, int out_size, void* d_ws, size_t ws_size,
                              hipStream_t stream) {
  const float* x  = (const float*)d_in[0];  // [512][256][512]
  const float* W  = (const float*)d_in[1];  // [1024][1536]
  const float* b  = (const float*)d_in[2];  // [1024]
  const float* Wc = (const float*)d_in[3];  // [1000][1024]
  const float* bc = (const float*)d_in[4];  // [1000]
  float* out = (float*)d_out;               // [256][1000]

  // Workspace layout (needs ~7.3 MB):
  //   W_bf   : 1024*1536 bf16   (3,145,728 B)
  //   Wc_bf  : 1008*1024 bf16   (2,064,384 B)
  //   h ping : 256*1024  f32    (1,048,576 B)
  //   h pong : 256*1024  f32    (1,048,576 B)
  char* ws = (char*)d_ws;
  __bf16* Wb  = (__bf16*)ws;
  __bf16* Wcb = (__bf16*)(ws + (size_t)HID * KCAT * 2);
  float*  h0  = (float*)(ws + (size_t)HID * KCAT * 2 + (size_t)NCLSP * HID * 2);
  float*  h1  = h0 + (size_t)BATCH * HID;

  cvt_f32_bf16<<<(HID * KCAT + 255) / 256, 256, 0, stream>>>(W, Wb, HID * KCAT);
  cvt_wc<<<(NCLSP * HID) / 256, 256, 0, stream>>>(Wc, Wcb);
  zero_f32<<<(BATCH * HID) / 256, 256, 0, stream>>>(h0, BATCH * HID);

  float* hp = h0;
  float* hn = h1;
  for (int t = 0; t < SEQ; ++t) {
    rnn_step<<<64, 256, 0, stream>>>(x + (size_t)t * BATCH * DIN, hp, Wb, b, hn);
    float* tmp = hp; hp = hn; hn = tmp;
  }
  classifier<<<126, 256, 0, stream>>>(hp, Wcb, bc, out);
}
